// selective_channel_aggregation_Module2_36301063586296
// MI455X (gfx1250) — matrix operand order repaired and checked by reading, not compiled, not run on hardware
//
#include <hip/hip_runtime.h>

// ---------------------------------------------------------------------------
// Problem constants (from reference): B=8, C=512, Q=512, H=W=64 -> N=4096
// ---------------------------------------------------------------------------
#define B_   8
#define C_   512
#define Q_   512
#define HW_  4096
#define LDSP 34          // padded LDS row stride (elements); 68B -> odd bank stride
#define LDSA_SZ (128 * LDSP)
#define LDSB_SZ (256 * LDSP)

typedef __attribute__((ext_vector_type(16))) __bf16 v16bf;
typedef __attribute__((ext_vector_type(8)))  float  v8f;

union FragBF { v16bf v; unsigned int u[8]; };

// ---------------------------------------------------------------------------
// fp32 -> bf16 conversion
// ---------------------------------------------------------------------------
__global__ void cvt_f32_to_bf16(const float* __restrict__ in,
                                __bf16* __restrict__ out, int n) {
  int i = blockIdx.x * blockDim.x + threadIdx.x;
  if (i < n) out[i] = (__bf16)in[i];
}

// Precompute BN scale/shift: y = x*scale + shift
__global__ void bn_prep(const float* __restrict__ w, const float* __restrict__ b,
                        const float* __restrict__ mean, const float* __restrict__ var,
                        float* __restrict__ scale, float* __restrict__ shift) {
  int i = blockIdx.x * blockDim.x + threadIdx.x;
  if (i < Q_) {
    float s = w[i] * rsqrtf(var[i] + 1e-5f);
    scale[i] = s;
    shift[i] = b[i] - mean[i] * s;
  }
}

// ---------------------------------------------------------------------------
// Staging helpers (256 threads). A tile: 128 rows x 32 k. B tile: 256 cols x 32 k.
// ---------------------------------------------------------------------------
__device__ __forceinline__ void gload_a(const __bf16* __restrict__ A, int lda,
                                        int k0, int tid, unsigned int ra[8]) {
  #pragma unroll
  for (int i = 0; i < 8; ++i) {
    int idx = tid + (i << 8);
    int r = idx >> 4, kd = (idx & 15) << 1;       // 16 dwords per row
    ra[i] = *(const unsigned int*)(A + (size_t)r * lda + k0 + kd);
  }
}
__device__ __forceinline__ void swrite_a(unsigned short* lA, int tid,
                                         const unsigned int ra[8]) {
  #pragma unroll
  for (int i = 0; i < 8; ++i) {
    int idx = tid + (i << 8);
    int r = idx >> 4, kd = (idx & 15) << 1;
    *(unsigned int*)&lA[r * LDSP + kd] = ra[i];
  }
}
// Direct: Bt stored [N x K] row-major, contiguous along k (pre-offset to col row)
__device__ __forceinline__ void gload_b_direct(const __bf16* __restrict__ Bt, int ldb,
                                               int k0, int tid, unsigned int rb[16]) {
  #pragma unroll
  for (int i = 0; i < 16; ++i) {
    int idx = tid + (i << 8);
    int r = idx >> 4, kd = (idx & 15) << 1;       // 256 rows x 16 dwords
    rb[i] = *(const unsigned int*)(Bt + (size_t)r * ldb + k0 + kd);
  }
}
__device__ __forceinline__ void swrite_b_direct(unsigned short* lB, int tid,
                                                const unsigned int rb[16]) {
  #pragma unroll
  for (int i = 0; i < 16; ++i) {
    int idx = tid + (i << 8);
    int r = idx >> 4, kd = (idx & 15) << 1;
    *(unsigned int*)&lB[r * LDSP + kd] = rb[i];
  }
}
// Transposed: B row-major [K x N]; read 32 k-rows x 256 n contiguous
__device__ __forceinline__ void gload_b_trans(const __bf16* __restrict__ B, int ldb,
                                              int k0, int tid, unsigned int rb[16]) {
  #pragma unroll
  for (int i = 0; i < 16; ++i) {
    int idx = tid + (i << 8);
    int kr = idx >> 7, nd = (idx & 127) << 1;     // 128 dwords per k-row
    rb[i] = *(const unsigned int*)(B + (size_t)(k0 + kr) * ldb + nd);
  }
}
__device__ __forceinline__ void swrite_b_trans(unsigned short* lB, int tid,
                                               const unsigned int rb[16]) {
  #pragma unroll
  for (int i = 0; i < 16; ++i) {
    int idx = tid + (i << 8);
    int kr = idx >> 7, nd = (idx & 127) << 1;
    lB[(nd    ) * LDSP + kr] = (unsigned short)(rb[i] & 0xffffu);
    lB[(nd + 1) * LDSP + kr] = (unsigned short)(rb[i] >> 16);
  }
}

// ---------------------------------------------------------------------------
// Double-buffered WMMA GEMM core: 128(M) x 256(N) block tile, K-step 32.
// 8 waves in 2(M) x 4(N); each wave owns 64x64 = 4x4 accumulators (16 WMMA/step).
// Pipeline: [swrite cur] -> sync -> [global load next into regs] -> [compute cur]
// ---------------------------------------------------------------------------
template<bool TRANSB>
__device__ __forceinline__ void gemm_tile_db(const __bf16* __restrict__ A, int lda,
                                             const __bf16* __restrict__ Bop, int ldb,
                                             int K,
                                             unsigned short* ldsA,   // [2][LDSA_SZ]
                                             unsigned short* ldsB,   // [2][LDSB_SZ]
                                             v8f acc[4][4]) {
  const int tid   = threadIdx.x;
  const int lane  = tid & 31;
  const int wv    = tid >> 5;
  const int wm    = wv >> 2;                 // 0..1
  const int wn    = wv & 3;                  // 0..3
  const int l15   = lane & 15;
  const int akoff = (lane < 16) ? 0 : 8;     // 16-bit A frag: K split at 8
  const int bkoff = (lane < 16) ? 0 : 8;     // 16-bit B frag: K split at 8

  unsigned int ra[8], rb[16];
  gload_a(A, lda, 0, tid, ra);
  if (TRANSB) gload_b_trans(Bop, ldb, 0, tid, rb);
  else        gload_b_direct(Bop, ldb, 0, tid, rb);

  int cur = 0;
  for (int k0 = 0; k0 < K; k0 += 32, cur ^= 1) {
    unsigned short* lA = ldsA + cur * LDSA_SZ;
    unsigned short* lB = ldsB + cur * LDSB_SZ;
    swrite_a(lA, tid, ra);
    if (TRANSB) swrite_b_trans(lB, tid, rb);
    else        swrite_b_direct(lB, tid, rb);
    __syncthreads();

    if (k0 + 32 < K) {          // prefetch next K-block while computing
      gload_a(A, lda, k0 + 32, tid, ra);
      if (TRANSB) gload_b_trans(Bop, ldb, k0 + 32, tid, rb);
      else        gload_b_direct(Bop, ldb, k0 + 32, tid, rb);
    }

    FragBF af[4], bfrag[4];
    #pragma unroll
    for (int mi = 0; mi < 4; ++mi) {
      int r = wm * 64 + mi * 16 + l15;
      #pragma unroll
      for (int v = 0; v < 8; ++v) {
        int kk = ((v & 3) << 1) + ((v >> 2) << 4) + akoff;
        af[mi].u[v] = *(const unsigned int*)&lA[r * LDSP + kk];
      }
    }
    #pragma unroll
    for (int ni = 0; ni < 4; ++ni) {
      int c = wn * 64 + ni * 16 + l15;
      #pragma unroll
      for (int v = 0; v < 8; ++v) {
        bfrag[ni].u[v] = *(const unsigned int*)&lB[c * LDSP + ((v & 3) << 1) + ((v >> 2) << 4) + bkoff];
      }
    }
    #pragma unroll
    for (int mi = 0; mi < 4; ++mi)
      #pragma unroll
      for (int ni = 0; ni < 4; ++ni)
        acc[mi][ni] = __builtin_amdgcn_wmma_f32_16x16x32_bf16(
            false, af[mi].v, false, bfrag[ni].v, (short)0, acc[mi][ni],
            false, false);
    __syncthreads();
  }
}

// ---------------------------------------------------------------------------
// GEMM1: q = ReLU(BN(conv_w @ x))   [per batch: (Q x C) * (C x N)]
// ---------------------------------------------------------------------------
__global__ __launch_bounds__(256)
void gemm1_conv_bn_relu(const __bf16* __restrict__ wb, const __bf16* __restrict__ xb,
                        const float* __restrict__ scale, const float* __restrict__ shift,
                        float* __restrict__ qf, __bf16* __restrict__ qb) {
  __shared__ alignas(16) unsigned short ldsA[2][LDSA_SZ];
  __shared__ alignas(16) unsigned short ldsB[2][LDSB_SZ];
  const int bz = blockIdx.z;
  const int rowBase = blockIdx.x * 128;
  const int colBase = blockIdx.y * 256;

  v8f acc[4][4] = {};
  const __bf16* A = wb + (size_t)rowBase * C_;
  const __bf16* B = xb + (size_t)bz * C_ * HW_ + colBase;   // [C x N], transposed stage
  gemm_tile_db<true>(A, C_, B, HW_, C_, &ldsA[0][0], &ldsB[0][0], acc);

  const int lane = threadIdx.x & 31, wv = threadIdx.x >> 5;
  const int wm = wv >> 2, wn = wv & 3, l15 = lane & 15;
  const int mh = (lane < 16) ? 0 : 8;
  #pragma unroll
  for (int mi = 0; mi < 4; ++mi)
    #pragma unroll
    for (int ni = 0; ni < 4; ++ni)
      #pragma unroll
      for (int vv = 0; vv < 8; ++vv) {
        int m = rowBase + wm * 64 + mi * 16 + mh + vv;
        int n = colBase + wn * 64 + ni * 16 + l15;
        float r = acc[mi][ni][vv] * scale[m] + shift[m];
        r = fmaxf(r, 0.0f);
        size_t o = ((size_t)bz * Q_ + m) * HW_ + n;
        qf[o] = r;
        qb[o] = (__bf16)r;
      }
}

// ---------------------------------------------------------------------------
// GEMM2: energy[b,q,c] = sum_n q[b,q,n] * x[b,c,n]   (K = N = 4096)
// ---------------------------------------------------------------------------
__global__ __launch_bounds__(256)
void gemm2_energy(const __bf16* __restrict__ qb, const __bf16* __restrict__ xb,
                  float* __restrict__ en) {
  __shared__ alignas(16) unsigned short ldsA[2][LDSA_SZ];
  __shared__ alignas(16) unsigned short ldsB[2][LDSB_SZ];
  const int bz = blockIdx.z;
  const int rowBase = blockIdx.x * 128;
  const int colBase = blockIdx.y * 256;

  v8f acc[4][4] = {};
  const __bf16* A  = qb + (size_t)bz * Q_ * HW_ + (size_t)rowBase * HW_;
  const __bf16* Bt = xb + (size_t)bz * C_ * HW_ + (size_t)colBase * HW_;
  gemm_tile_db<false>(A, HW_, Bt, HW_, HW_, &ldsA[0][0], &ldsB[0][0], acc);

  const int lane = threadIdx.x & 31, wv = threadIdx.x >> 5;
  const int wm = wv >> 2, wn = wv & 3, l15 = lane & 15;
  const int mh = (lane < 16) ? 0 : 8;
  #pragma unroll
  for (int mi = 0; mi < 4; ++mi)
    #pragma unroll
    for (int ni = 0; ni < 4; ++ni)
      #pragma unroll
      for (int vv = 0; vv < 8; ++vv) {
        int m = rowBase + wm * 64 + mi * 16 + mh + vv;
        int n = colBase + wn * 64 + ni * 16 + l15;
        en[((size_t)bz * Q_ + m) * C_ + n] = acc[mi][ni][vv];
      }
}

// ---------------------------------------------------------------------------
// Softmax over c of (max_c(e) - e) == exp(min_c(e) - e) / sum. One row/block.
// ---------------------------------------------------------------------------
__global__ __launch_bounds__(256)
void softmax_rows(const float* __restrict__ en, __bf16* __restrict__ at) {
  __shared__ float red[256];
  const int row = blockIdx.x;
  const int tid = threadIdx.x;
  const float* e = en + (size_t)row * C_;
  float v0 = e[tid], v1 = e[tid + 256];

  red[tid] = fminf(v0, v1);
  __syncthreads();
  for (int s = 128; s > 0; s >>= 1) {
    if (tid < s) red[tid] = fminf(red[tid], red[tid + s]);
    __syncthreads();
  }
  float mn = red[0];
  __syncthreads();

  float e0 = __expf(mn - v0), e1 = __expf(mn - v1);
  red[tid] = e0 + e1;
  __syncthreads();
  for (int s = 128; s > 0; s >>= 1) {
    if (tid < s) red[tid] += red[tid + s];
    __syncthreads();
  }
  float inv = 1.0f / red[0];
  at[(size_t)row * C_ + tid]       = (__bf16)(e0 * inv);
  at[(size_t)row * C_ + tid + 256] = (__bf16)(e1 * inv);
}

// ---------------------------------------------------------------------------
// GEMM3: out = gamma * (attention @ x) + q   [per batch: (Q x C) * (C x N)]
// ---------------------------------------------------------------------------
__global__ __launch_bounds__(256)
void gemm3_out(const __bf16* __restrict__ at, const __bf16* __restrict__ xb,
               const float* __restrict__ qf, const float* __restrict__ gamma,
               float* __restrict__ out) {
  __shared__ alignas(16) unsigned short ldsA[2][LDSA_SZ];
  __shared__ alignas(16) unsigned short ldsB[2][LDSB_SZ];
  const int bz = blockIdx.z;
  const int rowBase = blockIdx.x * 128;
  const int colBase = blockIdx.y * 256;

  v8f acc[4][4] = {};
  const __bf16* A = at + (size_t)bz * Q_ * C_ + (size_t)rowBase * C_;
  const __bf16* B = xb + (size_t)bz * C_ * HW_ + colBase;   // [C x N], transposed stage
  gemm_tile_db<true>(A, C_, B, HW_, C_, &ldsA[0][0], &ldsB[0][0], acc);

  const float g = gamma[0];
  const int lane = threadIdx.x & 31, wv = threadIdx.x >> 5;
  const int wm = wv >> 2, wn = wv & 3, l15 = lane & 15;
  const int mh = (lane < 16) ? 0 : 8;
  #pragma unroll
  for (int mi = 0; mi < 4; ++mi)
    #pragma unroll
    for (int ni = 0; ni < 4; ++ni)
      #pragma unroll
      for (int vv = 0; vv < 8; ++vv) {
        int m = rowBase + wm * 64 + mi * 16 + mh + vv;
        int n = colBase + wn * 64 + ni * 16 + l15;
        size_t o = ((size_t)bz * Q_ + m) * HW_ + n;
        out[o] = g * acc[mi][ni][vv] + qf[o];
      }
}

// ---------------------------------------------------------------------------
// Launch: x, conv_w, bn_w, bn_b, bn_mean, bn_var, gamma
// ---------------------------------------------------------------------------
extern "C" void kernel_launch(void* const* d_in, const int* in_sizes, int n_in,
                              void* d_out, int out_size, void* d_ws, size_t ws_size,
                              hipStream_t stream) {
  const float* x     = (const float*)d_in[0];
  const float* convw = (const float*)d_in[1];
  const float* bnw   = (const float*)d_in[2];
  const float* bnb   = (const float*)d_in[3];
  const float* bnm   = (const float*)d_in[4];
  const float* bnv   = (const float*)d_in[5];
  const float* gamma = (const float*)d_in[6];
  float* out = (float*)d_out;

  char* ws = (char*)d_ws;
  size_t o = 0;
  __bf16* xb = (__bf16*)(ws + o); o += (size_t)B_ * C_ * HW_ * 2;  // 33.5 MB
  __bf16* wb = (__bf16*)(ws + o); o += (size_t)Q_ * C_ * 2;        // 0.5 MB
  float*  qf = (float* )(ws + o); o += (size_t)B_ * Q_ * HW_ * 4;  // 67 MB
  __bf16* qb = (__bf16*)(ws + o); o += (size_t)B_ * Q_ * HW_ * 2;  // 33.5 MB
  float*  en = (float* )(ws + o); o += (size_t)B_ * Q_ * C_ * 4;   // 8.4 MB
  __bf16* at = (__bf16*)(ws + o); o += (size_t)B_ * Q_ * C_ * 2;   // 4.2 MB
  float* scale = (float*)(ws + o); o += (size_t)Q_ * 4;
  float* shift = (float*)(ws + o); o += (size_t)Q_ * 4;

  int nx = B_ * C_ * HW_;
  cvt_f32_to_bf16<<<(nx + 255) / 256, 256, 0, stream>>>(x, xb, nx);
  int nw = Q_ * C_;
  cvt_f32_to_bf16<<<(nw + 255) / 256, 256, 0, stream>>>(convw, wb, nw);
  bn_prep<<<2, 256, 0, stream>>>(bnw, bnb, bnm, bnv, scale, shift);

  gemm1_conv_bn_relu<<<dim3(Q_ / 128, HW_ / 256, B_), 256, 0, stream>>>(
      wb, xb, scale, shift, qf, qb);
  gemm2_energy<<<dim3(Q_ / 128, C_ / 256, B_), 256, 0, stream>>>(qb, xb, en);
  softmax_rows<<<B_ * Q_, 256, 0, stream>>>(en, at);
  gemm3_out<<<dim3(Q_ / 128, HW_ / 256, B_), 256, 0, stream>>>(
      at, xb, qf, gamma, out);
}
